// FlatLadder7x7NNUEv3_47519518163392
// MI455X (gfx1250) — compile-verified, matrix operand order checked by reading, not run on hardware
//
#include <hip/hip_runtime.h>
#include <hip/hip_bf16.h>

// ---------------------------------------------------------------------------
// Types
// ---------------------------------------------------------------------------
typedef __attribute__((ext_vector_type(16))) _Float16 half16;
typedef __attribute__((ext_vector_type(8)))  _Float16 half8;
typedef __attribute__((ext_vector_type(8)))  float    v8f;

union H16 { half16 v; half8 h[2]; };

#define QMAXF 0.9921875f   // 127/128

// ---------------------------------------------------------------------------
// Small device helpers
// ---------------------------------------------------------------------------
// mish(x) = x*tanh(softplus(x)) computed exactly without libm calls:
// e^{2*softplus(x)} = (1+e^x)^2  =>  mish = x*(t^2-1)/(t^2+1), t = 1+e^x.
// Clamp x before exp so t^2 stays finite (ratio -> 1 for large x).
__device__ inline float mishf(float x) {
    float ex = __expf(fminf(x, 20.f));
    float t  = 1.f + ex;
    float t2 = t * t;
    return x * (t2 - 1.f) * __builtin_amdgcn_rcpf(t2 + 1.f);
}

__device__ inline v8f vzero() {
    v8f z = {0.f,0.f,0.f,0.f,0.f,0.f,0.f,0.f};
    return z;
}

// Load a 16x32 A-fragment (M=items, K window) from LDS activations laid out
// [item][pos][128ch] (half). Lane l: row m=l&15, K-group per lane half.
__device__ inline half16 load_a(const _Float16* __restrict__ act, int P, int pos,
                                int kbase, int lane) {
    int m = lane & 15, grp = lane >> 4;
    const _Float16* b = act + (((m * P + pos) << 7) + kbase + (grp << 3));
    H16 r;
    r.h[0] = *(const half8*)(b);
    r.h[1] = *(const half8*)(b + 16);
    return r.v;
}

// Accumulate one 128-wide K block (4 WMMA k-steps) into c.
// pack: fragment-major packed f16 weights, fragment = (kw * nTiles + nt)
__device__ inline v8f gemm_acc(v8f c, const _Float16* __restrict__ in, int Pin, int pos,
                               const _Float16* __restrict__ pack, int kwBase,
                               int lane, int nt, int nTiles) {
#pragma unroll
    for (int kk = 0; kk < 4; ++kk) {
        half16 a = load_a(in, Pin, pos, kk << 5, lane);
        half16 b = *(const half16*)(pack + ((((kwBase + kk) * nTiles + nt) << 9) + (lane << 4)));
        c = __builtin_amdgcn_wmma_f32_16x16x32_f16(false, a, false, b, (short)0, c, false, false);
    }
    return c;
}

// scale/bias + mish, store to LDS activation buffer [item][Pout][128]
__device__ inline void store_mish(v8f c, const float* __restrict__ sb,
                                  _Float16* __restrict__ out, int Pout, int pos,
                                  int lane, int nt) {
    int n = (nt << 4) + (lane & 15);
    int grp = lane >> 4;
    float s = sb[n], bi = sb[128 + n];
#pragma unroll
    for (int r = 0; r < 8; ++r) {
        int m = r + (grp << 3);
        out[((m * Pout + pos) << 7) + n] = (_Float16)mishf(c[r] * s + bi);
    }
}

// scale/bias + clip[-1,qmax] + round(x*128)/128, atomically accumulate to f32 feature
__device__ inline void store_quant_atomic(v8f c, const float* __restrict__ sb,
                                          float* __restrict__ dst, int b0, int chDim,
                                          int lane, int nt) {
    int n = (nt << 4) + (lane & 15);
    int grp = lane >> 4;
    float s = sb[n], bi = sb[128 + n];
#pragma unroll
    for (int r = 0; r < 8; ++r) {
        int m = r + (grp << 3);
        float y = c[r] * s + bi;
        y = fminf(fmaxf(y, -1.f), QMAXF);
        y = rintf(y * 128.f) * 0.0078125f;
        __hip_atomic_fetch_add(dst + (size_t)(b0 + m) * chDim + n, y,
                               __ATOMIC_RELAXED, __HIP_MEMORY_SCOPE_AGENT);
    }
}

// ---------------------------------------------------------------------------
// Prep kernels
// ---------------------------------------------------------------------------

// Fold BN: scale = g*rsqrt(v+eps); bias = (bconv - m)*scale + b.
// Output slot: out[0..N) = scale, out[128..128+N) = bias.
__global__ void k_sb(const float* g, const float* b, const float* m, const float* v,
                     const float* bconv, float* out, int N) {
    int i = blockIdx.x * blockDim.x + threadIdx.x;
    if (i < N) {
        float s = g[i] * rsqrtf(v[i] + 1e-5f);
        out[i]       = s;
        out[128 + i] = (bconv[i] - m[i]) * s + b[i];
    }
}

// Pack f32 weights into f16 WMMA-B fragment layout.
// Fragment = 512 halfs; lane l owns 16 contiguous halfs; (n,k) mapping mirrors
// the A-fragment K interleave (groups {0..7,16..23} / {8..15,24..31}).
// mode 0: w is [N][K] row-major.
// mode 1: ladder lw (3,N,Cin): k -> t=k/Cin, i=k%Cin, w[(t*N+n)*Cin+i].
// mode 2: conv2x2 w (N,Cin,2,2): k -> p=k/Cin (p=dy*2+dx), i=k%Cin.
__global__ void k_pack(const float* __restrict__ w, _Float16* __restrict__ out,
                       int N, int K, int mode, int Cin) {
    int total = N * K;
    int nTiles = N >> 4;
    for (int idx = blockIdx.x * blockDim.x + threadIdx.x; idx < total;
         idx += gridDim.x * blockDim.x) {
        int frag = idx >> 9;
        int r = idx & 511;
        int l = r >> 4, j = r & 15;
        int kw = frag / nTiles, nt = frag - kw * nTiles;
        int n = (nt << 4) + (l & 15);
        int grp = l >> 4;
        int k = (kw << 5) + ((j < 8) ? (grp * 8 + j) : (16 + grp * 8 + (j - 8)));
        float val;
        if (mode == 0) {
            val = w[n * K + k];
        } else if (mode == 1) {
            int t = k / Cin, i = k - t * Cin;
            val = w[(t * N + n) * Cin + i];
        } else {
            int p = k / Cin, i = k - p * Cin;
            int dy = p >> 1, dx = p & 1;
            val = w[((n * Cin + i) * 2 + dy) * 2 + dx];
        }
        out[idx] = (_Float16)val;
    }
}

// fake quant: round(w*s)/s
__global__ void k_fq(const float* __restrict__ w, float* __restrict__ out, int n, float s) {
    int i = blockIdx.x * blockDim.x + threadIdx.x;
    if (i < n) out[i] = rintf(w[i] * s) / s;
}

__global__ void k_zero(float* p, long n) {
    for (long i = blockIdx.x * (long)blockDim.x + threadIdx.x; i < n;
         i += (long)gridDim.x * blockDim.x)
        p[i] = 0.f;
}

// ---------------------------------------------------------------------------
// Ladder mapping kernel (m1: 8 variants, m2: 4 variants)
// One 256-thread workgroup = 8 waves handles a 16-item tile of one variant.
// Packed weight offsets (halfs): L1=0, L2=16384, L3=65536, L4=81920,
// L5=131072, L6=147456.  Scale/bias slots: layer*256 floats.
// ---------------------------------------------------------------------------
struct LadderArgs {
    const float*    x;     // (B,2,7,7)
    const float*    lw0;   // stage0 ladder weights (3,128,2) raw f32
    const float*    sb;    // 7 slots * 256 floats
    const _Float16* pack;  // packed L1..L6
    float*          gf;    // (B,128) accumulator
    int             nb;    // batch
    unsigned char   tbl[8][16];
};

__global__ __launch_bounds__(256) void k_ladder(LadderArgs A) {
    __shared__ __align__(32) _Float16 shA[16 * 9 * 128]; // L0 out (9 pos)
    __shared__ __align__(32) _Float16 shT[16 * 128];     // per-position temp
    __shared__ __align__(32) _Float16 shB[16 * 4 * 128]; // L2 out (4 pos); aliases xs
    __shared__ __align__(32) _Float16 shC[16 * 128];     // L4 out; aliases lw0 stage

    float* xs   = (float*)shB; // [16][2][16], dead before shB is written
    float* lw0s = (float*)shC; // 768 floats, dead before shC is written

    int tiles = A.nb >> 4;
    int v  = blockIdx.x / tiles;
    int b0 = (blockIdx.x - v * tiles) << 4;
    int tid = threadIdx.x, lane = tid & 31, wv = tid >> 5;

    // --- stage-0 weights to LDS + gather rotated/transposed 4x4 crop ---
    for (int i = tid; i < 768; i += 256) lw0s[i] = A.lw0[i];
    for (int i = tid; i < 512; i += 256) {
        int p = i & 15, c = (i >> 4) & 1, it = i >> 5;
        xs[(it * 2 + c) * 16 + p] =
            A.x[((size_t)(b0 + it) * 2 + c) * 49 + A.tbl[v][p]];
    }
    __syncthreads();

    // --- L0: ladder conv 4x4->3x3, din=2 (scalar) + bn + mish ---
    for (int i = tid; i < 16 * 9 * 128; i += 256) {
        int o = i & 127, r = i >> 7, p = r % 9, it = r / 9;
        int py = p / 3, px = p - py * 3;
        int p0 = py * 4 + px, p1 = p0 + 4, p2 = p0 + 5;
        float acc = 0.f;
#pragma unroll
        for (int c = 0; c < 2; ++c) {
            acc += lw0s[(0 * 128 + o) * 2 + c] * xs[(it * 2 + c) * 16 + p0];
            acc += lw0s[(1 * 128 + o) * 2 + c] * xs[(it * 2 + c) * 16 + p1];
            acc += lw0s[(2 * 128 + o) * 2 + c] * xs[(it * 2 + c) * 16 + p2];
        }
        shA[((it * 9 + p) << 7) + o] = (_Float16)mishf(acc * A.sb[o] + A.sb[128 + o]);
    }
    __syncthreads();

    // --- stage1 fused: L1 (1x1) per needed position -> L2 (ladder K=384) ---
    v8f acc2[4];
#pragma unroll
    for (int p = 0; p < 4; ++p) acc2[p] = vzero();

    const int QL[8] = {0, 1, 3, 4, 5, 6, 7, 8}; // pos 2 never consumed
    for (int qi = 0; qi < 8; ++qi) {
        int q = QL[qi];
        v8f c1 = gemm_acc(vzero(), shA, 9, q, A.pack + 0, 0, lane, wv, 8);
        __syncthreads();                        // prior consumers of shT done
        store_mish(c1, A.sb + 256, shT, 1, 0, lane, wv);
        __syncthreads();
#pragma unroll
        for (int p = 0; p < 4; ++p) {
            int py = p >> 1, px = p & 1;
            int pin0 = py * 3 + px, pin1 = pin0 + 3, pin2 = pin0 + 4;
            if (pin0 == q) acc2[p] = gemm_acc(acc2[p], shT, 1, 0, A.pack + 16384, 0, lane, wv, 8);
            if (pin1 == q) acc2[p] = gemm_acc(acc2[p], shT, 1, 0, A.pack + 16384, 4, lane, wv, 8);
            if (pin2 == q) acc2[p] = gemm_acc(acc2[p], shT, 1, 0, A.pack + 16384, 8, lane, wv, 8);
        }
    }
    __syncthreads();
#pragma unroll
    for (int p = 0; p < 4; ++p) store_mish(acc2[p], A.sb + 2 * 256, shB, 4, p, lane, wv);
    __syncthreads();

    // --- stage2 fused: L3 (1x1) per needed position -> L4 (ladder K=384) ---
    v8f acc4 = vzero();
    const int Q2[3] = {0, 2, 3}; // taps (0,0),(1,0),(1,1) on 2x2 grid
    for (int qi = 0; qi < 3; ++qi) {
        v8f c3 = gemm_acc(vzero(), shB, 4, Q2[qi], A.pack + 65536, 0, lane, wv, 8);
        __syncthreads();
        store_mish(c3, A.sb + 3 * 256, shT, 1, 0, lane, wv);
        __syncthreads();
        acc4 = gemm_acc(acc4, shT, 1, 0, A.pack + 81920, qi << 2, lane, wv, 8);
    }
    __syncthreads();
    store_mish(acc4, A.sb + 4 * 256, shC, 1, 0, lane, wv);
    __syncthreads();

    // --- L5: 1x1 + mish ---
    v8f c5 = gemm_acc(vzero(), shC, 1, 0, A.pack + 131072, 0, lane, wv, 8);
    store_mish(c5, A.sb + 5 * 256, shT, 1, 0, lane, wv);
    __syncthreads();

    // --- L6: final 1x1 + bn + clip + quant -> atomic accumulate ---
    v8f c6 = gemm_acc(vzero(), shT, 1, 0, A.pack + 147456, 0, lane, wv, 8);
    store_quant_atomic(c6, A.sb + 6 * 256, A.gf, b0, 128, lane, wv);
}

// ---------------------------------------------------------------------------
// Conv3x3 mapping kernel: 25 crop variants over 6 nets.
// Packed per-net stride = 86016 halfs: C1@0 (K=512), C2@65536, C3@81920 (N=32).
// Scale/bias per-net stride = 1024 floats (4 slots).
// ---------------------------------------------------------------------------
struct ConvArgs {
    const float*    x;
    const float*    w1[6];   // raw (128,2,2,2)
    const float*    sb;      // base, per-net stride 1024 floats
    const _Float16* pack;    // base, per-net stride 86016 halfs
    float*          cf;      // (B,32) accumulator
    int             nb;
    unsigned char   tbl[25][9];
    unsigned char   ci[25];
};

__global__ __launch_bounds__(256) void k_conv(ConvArgs A) {
    __shared__ __align__(32) _Float16 sh0[16 * 4 * 128];
    __shared__ __align__(32) _Float16 sh1[16 * 128];     // aliases w1 staging
    __shared__ __align__(32) _Float16 sh2[16 * 128];
    __shared__ float xs[16 * 2 * 9];

    float* w1s = (float*)sh1; // 1024 floats, dead before sh1 is written

    int tiles = A.nb >> 4;
    int v  = blockIdx.x / tiles;
    int b0 = (blockIdx.x - v * tiles) << 4;
    int tid = threadIdx.x, lane = tid & 31, wv = tid >> 5;

    int ci = A.ci[v];
    const _Float16* pack = A.pack + (size_t)ci * 86016;
    const float*    sb   = A.sb + ci * 1024;
    const float*    w1   = A.w1[ci];

    for (int i = tid; i < 1024; i += 256) w1s[i] = w1[i];
    for (int i = tid; i < 288; i += 256) {
        int p = i % 9, rest = i / 9;
        int c = rest & 1, it = rest >> 1;
        xs[(it * 2 + c) * 9 + p] =
            A.x[((size_t)(b0 + it) * 2 + c) * 49 + A.tbl[v][p]];
    }
    __syncthreads();

    // C0: full 2x2 conv 3x3->2x2, din=2 (scalar) + bn + mish
    for (int i = tid; i < 16 * 4 * 128; i += 256) {
        int o = i & 127, r = i >> 7, p = r & 3, it = r >> 2;
        int py = p >> 1, px = p & 1;
        float acc = 0.f;
#pragma unroll
        for (int ii = 0; ii < 2; ++ii)
#pragma unroll
            for (int dy = 0; dy < 2; ++dy)
#pragma unroll
                for (int dx = 0; dx < 2; ++dx)
                    acc += w1s[((o * 2 + ii) * 2 + dy) * 2 + dx] *
                           xs[(it * 2 + ii) * 9 + (py + dy) * 3 + (px + dx)];
        sh0[((it * 4 + p) << 7) + o] = (_Float16)mishf(acc * sb[o] + sb[128 + o]);
    }
    __syncthreads();

    // C1: 2x2 conv as K=512 GEMM over the 4 positions
    v8f c = vzero();
#pragma unroll
    for (int t = 0; t < 4; ++t)
        c = gemm_acc(c, sh0, 4, t, pack + 0, t << 2, lane, wv, 8);
    store_mish(c, sb + 256, sh1, 1, 0, lane, wv);
    __syncthreads();

    // C2: 1x1
    c = gemm_acc(vzero(), sh1, 1, 0, pack + 65536, 0, lane, wv, 8);
    store_mish(c, sb + 512, sh2, 1, 0, lane, wv);
    __syncthreads();

    // C3: 1x1 to 32 channels + clip + quant -> atomic accumulate
    if (wv < 2) {
        c = gemm_acc(vzero(), sh2, 1, 0, pack + 81920, 0, lane, wv, 2);
        store_quant_atomic(c, sb + 768, A.cf, b0, 32, lane, wv);
    }
}

// ---------------------------------------------------------------------------
// Value head: feature(160) -> 32 -> 32 -> 3 with clipped inputs; zero policy
// ---------------------------------------------------------------------------
__device__ inline float fclamp(float x, float lo) { return fminf(fmaxf(x, lo), QMAXF); }

__global__ __launch_bounds__(256) void k_value(const float* __restrict__ gf,
                                               const float* __restrict__ cf,
                                               const float* __restrict__ w0, const float* __restrict__ b0,
                                               const float* __restrict__ w1, const float* __restrict__ b1,
                                               const float* __restrict__ w2, const float* __restrict__ b2,
                                               float* __restrict__ out, int B) {
    __shared__ float hsh[256 * 33];
    int tid = threadIdx.x;
    int row = blockIdx.x * 256 + tid;
    if (row < B) {
        float* h = hsh + tid * 33;
        for (int o = 0; o < 32; ++o) {
            float s = b0[o];
            const float* wr = w0 + o * 160;
            for (int i = 0; i < 128; ++i) s += wr[i] * fclamp(gf[(size_t)row * 128 + i], -1.f);
            for (int i = 0; i < 32; ++i)  s += wr[128 + i] * fclamp(cf[(size_t)row * 32 + i], -1.f);
            h[o] = fclamp(s, 0.f);
        }
        float v0 = b2[0], v1 = b2[1], v2 = b2[2];
        for (int o = 0; o < 32; ++o) {
            float s = b1[o];
            const float* wr = w1 + o * 32;
            for (int i = 0; i < 32; ++i) s += wr[i] * h[i];
            s = fclamp(s, 0.f);
            v0 += w2[0 * 32 + o] * s;
            v1 += w2[1 * 32 + o] * s;
            v2 += w2[2 * 32 + o] * s;
        }
        out[(size_t)row * 3 + 0] = v0;
        out[(size_t)row * 3 + 1] = v1;
        out[(size_t)row * 3 + 2] = v2;
    }
    // policy = zeros
    for (long i = blockIdx.x * 256L + tid; i < (long)B * 49; i += (long)gridDim.x * 256)
        out[(long)B * 3 + i] = 0.f;
}

// ---------------------------------------------------------------------------
// Host side
// ---------------------------------------------------------------------------

// rot90 (counter-clockwise, jnp semantics) / transpose applied to crop index grids
static void build_tbl(unsigned char* out, int N, int y0, int x0, int k, bool t) {
    int a[4][4], b[4][4];
    for (int y = 0; y < N; ++y)
        for (int x = 0; x < N; ++x) a[y][x] = (y0 + y) * 7 + (x0 + x);
    if (t) {
        for (int y = 0; y < N; ++y)
            for (int x = 0; x < N; ++x) b[y][x] = a[x][y];
        for (int y = 0; y < N; ++y)
            for (int x = 0; x < N; ++x) a[y][x] = b[y][x];
    }
    int kk = ((k % 4) + 4) % 4;
    for (int r = 0; r < kk; ++r) {
        for (int y = 0; y < N; ++y)
            for (int x = 0; x < N; ++x) b[y][x] = a[x][N - 1 - y];
        for (int y = 0; y < N; ++y)
            for (int x = 0; x < N; ++x) a[y][x] = b[y][x];
    }
    for (int y = 0; y < N; ++y)
        for (int x = 0; x < N; ++x) out[y * N + x] = (unsigned char)a[y][x];
}

// (y0,x0,rot,transpose)
static const int M1T[8][4] = {{0,0,0,0},{0,0,0,1},{0,3,1,0},{0,3,-1,1},
                              {3,3,2,0},{3,3,2,1},{3,0,-1,0},{3,0,1,1}};
static const int M2T[4][3] = {{0,0,1},{0,3,2},{3,3,-1},{3,0,0}};
// 25 crops: which net / rot / transpose
static const int CC[25] = {0,1,2,1,0, 1,3,4,3,1, 2,4,5,4,2, 1,3,4,3,1, 0,1,2,1,0};
static const int CK[25] = {0,0,0,-1,-1, 0,0,0,-1,-1, -1,-1,0,1,1, -1,-1,2,2,2, -1,1,2,2,2};
static const int CT[25] = {0,0,0,1,0, 0,0,0,0,1, 0,0,0,0,0, 0,0,0,0,1, 0,1,0,0,0};

extern "C" void kernel_launch(void* const* d_in, const int* in_sizes, int n_in,
                              void* d_out, int out_size, void* d_ws, size_t ws_size,
                              hipStream_t stream) {
    (void)n_in; (void)out_size; (void)ws_size;
    const float* x = (const float*)d_in[0];
    const int B = in_sizes[0] / 98;            // (B,2,7,7)
    const int tiles = B / 16;

    // --- input index map (tree-flatten, dict keys sorted) ---
    // d_in[0]=x; convs[i] base 1+i*24: b1,b2,b3,b4, bn1(g,b,m,v), bn2(..), bn3(..),
    //   bn4(..), w1,w2,w3,w4.  m1@145 / m2@187: fb, fbn(g,b,m,v), fw, stages[3]
    //   (cb, cbn(g,b,m,v), cw, lb, lbn(g,b,m,v), lw).  vl@229: w0,b0,w1,b1,w2,b2.
    const int M1 = 145, M2 = 187, VL = 229;

    // --- workspace layout ---
    float*     gf     = (float*)d_ws;                       // B*128
    float*     cf     = gf + (size_t)B * 128;               // B*32
    _Float16*  m1pack = (_Float16*)(cf + (size_t)B * 32);   // 163840 halfs
    _Float16*  m2pack = m1pack + 163840;
    _Float16*  cpack  = m2pack + 163840;                    // 6*86016 halfs
    float*     m1sb   = (float*)(cpack + 6 * 86016);        // 7*256
    float*     m2sb   = m1sb + 7 * 256;
    float*     csb    = m2sb + 7 * 256;                     // 6*1024
    float*     vw0    = csb + 6 * 1024;                     // fq'd value head
    float*     vb0    = vw0 + 5120;
    float*     vw1    = vb0 + 32;
    float*     vb1    = vw1 + 1024;
    float*     vw2    = vb1 + 32;
    float*     vb2    = vw2 + 96;

    auto F = [&](int i) { return (const float*)d_in[i]; };
    auto SB = [&](int g, int b, int m, int v, int bc, float* out, int N) {
        k_sb<<<1, 128, 0, stream>>>(F(g), F(b), F(m), F(v), F(bc), out, N);
    };
    auto PACK = [&](int wi, _Float16* out, int N, int K, int mode, int Cin) {
        int total = N * K;
        k_pack<<<(total + 255) / 256, 256, 0, stream>>>(F(wi), out, N, K, mode, Cin);
    };

    // --- ladder prep (m1 and m2) ---
    const int mbase[2] = {M1, M2};
    _Float16* mpk[2] = {m1pack, m2pack};
    float*    msb[2] = {m1sb, m2sb};
    for (int mm = 0; mm < 2; ++mm) {
        int m = mbase[mm];
        int st0 = m + 6, st1 = m + 18, st2 = m + 30;
        SB(st0 + 7, st0 + 8, st0 + 9, st0 + 10, st0 + 6, msb[mm] + 0,    128); // L0 (lbn,lb)
        SB(st0 + 1, st0 + 2, st0 + 3, st0 + 4,  st0 + 0, msb[mm] + 256,  128); // L1 (cbn,cb)
        SB(st1 + 7, st1 + 8, st1 + 9, st1 + 10, st1 + 6, msb[mm] + 512,  128); // L2
        SB(st1 + 1, st1 + 2, st1 + 3, st1 + 4,  st1 + 0, msb[mm] + 768,  128); // L3
        SB(st2 + 7, st2 + 8, st2 + 9, st2 + 10, st2 + 6, msb[mm] + 1024, 128); // L4
        SB(st2 + 1, st2 + 2, st2 + 3, st2 + 4,  st2 + 0, msb[mm] + 1280, 128); // L5
        SB(m + 1,   m + 2,   m + 3,   m + 4,    m + 0,   msb[mm] + 1536, 128); // L6 (fbn,fb)
        PACK(st0 + 5,  mpk[mm] + 0,      128, 128, 0, 128); // L1 cw
        PACK(st1 + 11, mpk[mm] + 16384,  128, 384, 1, 128); // L2 lw
        PACK(st1 + 5,  mpk[mm] + 65536,  128, 128, 0, 128); // L3 cw
        PACK(st2 + 11, mpk[mm] + 81920,  128, 384, 1, 128); // L4 lw
        PACK(st2 + 5,  mpk[mm] + 131072, 128, 128, 0, 128); // L5 cw
        PACK(m + 5,    mpk[mm] + 147456, 128, 128, 0, 128); // L6 fw
    }

    // --- conv prep ---
    for (int i = 0; i < 6; ++i) {
        int cb = 1 + i * 24;
        SB(cb + 4,  cb + 5,  cb + 6,  cb + 7,  cb + 0, csb + i * 1024 + 0,   128); // C0
        SB(cb + 8,  cb + 9,  cb + 10, cb + 11, cb + 1, csb + i * 1024 + 256, 128); // C1
        SB(cb + 12, cb + 13, cb + 14, cb + 15, cb + 2, csb + i * 1024 + 512, 128); // C2
        SB(cb + 16, cb + 17, cb + 18, cb + 19, cb + 3, csb + i * 1024 + 768, 32);  // C3
        PACK(cb + 21, cpack + (size_t)i * 86016 + 0,     128, 512, 2, 128); // w2
        PACK(cb + 22, cpack + (size_t)i * 86016 + 65536, 128, 128, 0, 128); // w3
        PACK(cb + 23, cpack + (size_t)i * 86016 + 81920, 32,  128, 0, 128); // w4
    }

    // --- value head fq ---
    k_fq<<<(5120 + 255) / 256, 256, 0, stream>>>(F(VL + 0), vw0, 5120, 128.f);
    k_fq<<<1, 64, 0, stream>>>(F(VL + 1), vb0, 32, 16384.f);
    k_fq<<<(1024 + 255) / 256, 256, 0, stream>>>(F(VL + 2), vw1, 1024, 128.f);
    k_fq<<<1, 64, 0, stream>>>(F(VL + 3), vb1, 32, 16384.f);
    k_fq<<<1, 128, 0, stream>>>(F(VL + 4), vw2, 96, 128.f);
    k_fq<<<1, 64, 0, stream>>>(F(VL + 5), vb2, 3, 16384.f);

    // --- zero feature accumulators ---
    k_zero<<<256, 256, 0, stream>>>(gf, (long)B * 160);

    // --- mapping1 ---
    {
        LadderArgs A = {};
        A.x = x;
        A.lw0  = F(M1 + 6 + 11);
        A.sb   = m1sb;
        A.pack = m1pack;
        A.gf   = gf;
        A.nb   = B;
        for (int v = 0; v < 8; ++v)
            build_tbl(&A.tbl[v][0], 4, M1T[v][0], M1T[v][1], M1T[v][2], M1T[v][3] != 0);
        k_ladder<<<8 * tiles, 256, 0, stream>>>(A);
    }
    // --- mapping2 ---
    {
        LadderArgs A = {};
        A.x = x;
        A.lw0  = F(M2 + 6 + 11);
        A.sb   = m2sb;
        A.pack = m2pack;
        A.gf   = gf;
        A.nb   = B;
        for (int v = 0; v < 4; ++v)
            build_tbl(&A.tbl[v][0], 4, M2T[v][0], M2T[v][1], M2T[v][2], false);
        k_ladder<<<4 * tiles, 256, 0, stream>>>(A);
    }
    // --- conv crops ---
    {
        ConvArgs A = {};
        A.x = x;
        for (int i = 0; i < 6; ++i) A.w1[i] = F(1 + i * 24 + 20);
        A.sb   = csb;
        A.pack = cpack;
        A.cf   = cf;
        A.nb   = B;
        for (int v = 0; v < 25; ++v) {
            int y = v / 5, xx = v % 5;
            build_tbl(&A.tbl[v][0], 3, y, xx, CK[v], CT[v] != 0);
            A.ci[v] = (unsigned char)CC[v];
        }
        k_conv<<<25 * tiles, 256, 0, stream>>>(A);
    }
    // --- value head + policy zeros ---
    k_value<<<(B + 255) / 256, 256, 0, stream>>>(gf, cf, vw0, vb0, vw1, vb1, vw2, vb2,
                                                 (float*)d_out, B);
}
